// GraphConvolution_18056042512529
// MI455X (gfx1250) — matrix operand order repaired and checked by reading, not compiled, not run on hardware
//
#include <hip/hip_runtime.h>
#include <math.h>

#define BB  64
#define EE  4
#define NN  512
#define DIN 64
#define DD0 64
#define DD1 32

typedef __attribute__((ext_vector_type(16))) __bf16 v16bf;
typedef __attribute__((ext_vector_type(8)))  float  v8f;

// ---------------------------------------------------------------------------
// K0: inv_deg[b,e,n] = 1 / (sum_m adj[b,e,n,m] + 1)   -- one wave per row
// ---------------------------------------------------------------------------
__global__ __launch_bounds__(256) void k_invdeg(const float* __restrict__ adj,
                                                float* __restrict__ invdeg) {
  const int lane = threadIdx.x & 31;
  const int row  = blockIdx.x * 8 + (threadIdx.x >> 5);   // [0, BB*EE*NN)
  const float* p = adj + (size_t)row * NN;
  float s = 0.f;
  #pragma unroll
  for (int j = 0; j < NN / 32; ++j) s += p[lane + 32 * j];
  #pragma unroll
  for (int m = 16; m >= 1; m >>= 1) s += __shfl_xor(s, m, 32);
  if (lane == 0) invdeg[row] = 1.0f / (s + 1.0f);
}

// ---------------------------------------------------------------------------
// K1: h0 = x @ W1 + b1  (f32), plus transposed bf16 copy h0T[b][d][n]
// ---------------------------------------------------------------------------
__global__ __launch_bounds__(256) void k_lin1(const float* __restrict__ x,
                                              const float* __restrict__ W1,
                                              const float* __restrict__ b1,
                                              float* __restrict__ h0,
                                              __bf16* __restrict__ h0T) {
  const int idx = blockIdx.x * 256 + threadIdx.x;   // BB*NN*DD0 = 2^21
  const int d = idx & (DD0 - 1);
  const int n = (idx >> 6) & (NN - 1);
  const int b = idx >> 15;
  const float* xr = x + ((size_t)b * NN + n) * DIN;
  float acc = b1[d];
  #pragma unroll 8
  for (int k = 0; k < DIN; ++k) acc = fmaf(xr[k], W1[k * DD0 + d], acc);
  h0[idx] = acc;
  h0T[((size_t)b * DD0 + d) * NN + n] = (__bf16)acc;
}

// ---------------------------------------------------------------------------
// K3: o0 = hidden @ W2 + b2 (f32), plus transposed bf16 copy o0T[b][d][n]
// ---------------------------------------------------------------------------
__global__ __launch_bounds__(256) void k_lin2(const float* __restrict__ hidden,
                                              const float* __restrict__ W2,
                                              const float* __restrict__ b2,
                                              float* __restrict__ o0,
                                              __bf16* __restrict__ o0T) {
  const int idx = blockIdx.x * 256 + threadIdx.x;   // BB*NN*DD1 = 2^20
  const int d = idx & (DD1 - 1);
  const int n = (idx >> 5) & (NN - 1);
  const int b = idx >> 14;
  const float* hr = hidden + ((size_t)b * NN + n) * DD0;
  float acc = b2[d];
  #pragma unroll 8
  for (int k = 0; k < DD0; ++k) acc = fmaf(hr[k], W2[k * DD1 + d], acc);
  o0[idx] = acc;
  o0T[((size_t)b * DD1 + d) * NN + n] = (__bf16)acc;
}

// ---------------------------------------------------------------------------
// K2/K4: out[b,rows,:] = tanh( sum_e (inv_deg .* adj[b,e,rows,:]) @ self[b]
//                             + self[b,rows,:] )
// One wave computes a 16-row tile x (NT*16) columns with bf16 WMMA, f32 accum.
//   A tile 16x32  : adj rows, scaled + cvt to bf16 (contiguous k order, matches B)
//   B tile 32x16  : contiguous 32B/lane load from transposed bf16 operand
// ---------------------------------------------------------------------------
template <int NT>
__global__ __launch_bounds__(256) void k_agg(const float* __restrict__ adj,
                                             const float* __restrict__ invdeg,
                                             const float* __restrict__ self,   // [B][N][NT*16] f32
                                             const __bf16* __restrict__ selfT, // [B][NT*16][N] bf16
                                             float* __restrict__ outF) {       // [B][N][NT*16] f32
  const int lane = threadIdx.x & 31;
  const int tile = blockIdx.x * 8 + (threadIdx.x >> 5);  // [0, BB*NN/16)
  const int b    = tile >> 5;                            // 32 row-tiles per batch
  const int row0 = (tile & 31) << 4;
  const int hlf  = lane >> 4;                            // 0 or 1
  const int r    = lane & 15;
  const int D    = NT * 16;

  const float*  adj_b = adj    + (size_t)b * EE * NN * NN;
  const float*  inv_b = invdeg + (size_t)b * EE * NN;
  const __bf16* bT    = selfT  + (size_t)b * D * NN;

  // per-lane row scale for each edge type (A-lane holds row r of the tile)
  float sc[EE];
  #pragma unroll
  for (int e = 0; e < EE; ++e) sc[e] = inv_b[e * NN + row0 + r];

  v8f c[NT];
  #pragma unroll
  for (int nt = 0; nt < NT; ++nt)
    #pragma unroll
    for (int i = 0; i < 8; ++i) c[nt][i] = 0.0f;

  for (int k0 = 0; k0 < NN; k0 += 32) {
    // B tiles: lane = column (lane&15), K = k0 + 16*hlf .. +15 contiguous
    v16bf bt[NT];
    #pragma unroll
    for (int nt = 0; nt < NT; ++nt)
      bt[nt] = *(const v16bf*)(bT + (size_t)(nt * 16 + r) * NN + (k0 + hlf * 16));

    #pragma unroll
    for (int e = 0; e < EE; ++e) {
      // A tile: lane row = r; elems 0..15 = K k0+16*hlf .. +15 contiguous
      const float* ap = adj_b + ((size_t)e * NN + row0 + r) * NN + (k0 + hlf * 16);
      v8f lo = *(const v8f*)(ap);
      v8f hi = *(const v8f*)(ap + 8);
      v16bf a;
      #pragma unroll
      for (int i = 0; i < 8; ++i) {
        a[i]     = (__bf16)(lo[i] * sc[e]);
        a[i + 8] = (__bf16)(hi[i] * sc[e]);
      }
      #pragma unroll
      for (int nt = 0; nt < NT; ++nt)
        c[nt] = __builtin_amdgcn_wmma_f32_16x16x32_bf16(
            false, a, false, bt[nt], (short)0, c[nt], false, false);
    }
  }

  // C/D layout: vgpr v, lane -> (M = v + 8*hlf, col = r)
  const float* self_b = self + (size_t)b * NN * D;
  float*       out_b  = outF + (size_t)b * NN * D;
  #pragma unroll
  for (int nt = 0; nt < NT; ++nt) {
    const int d = nt * 16 + r;
    #pragma unroll
    for (int v = 0; v < 8; ++v) {
      const int m = row0 + v + 8 * hlf;
      out_b[(size_t)m * D + d] = tanhf(c[nt][v] + self_b[(size_t)m * D + d]);
    }
  }
}

// ---------------------------------------------------------------------------
extern "C" void kernel_launch(void* const* d_in, const int* in_sizes, int n_in,
                              void* d_out, int out_size, void* d_ws, size_t ws_size,
                              hipStream_t stream) {
  const float* x   = (const float*)d_in[0];
  const float* adj = (const float*)d_in[1];
  const float* W1  = (const float*)d_in[2];
  const float* b1  = (const float*)d_in[3];
  const float* W2  = (const float*)d_in[4];
  const float* b2  = (const float*)d_in[5];
  float* out = (float*)d_out;

  char* ws = (char*)d_ws;
  float*  invdeg = (float*)(ws);                                   // 512 KB
  float*  h0     = (float*)(ws + (512 << 10));                     // 8 MB
  __bf16* h0T    = (__bf16*)(ws + (512 << 10) + (8 << 20));        // 4 MB
  float*  hidden = (float*)(ws + (512 << 10) + (12 << 20));        // 8 MB
  float*  o0     = (float*)(ws + (512 << 10) + (20 << 20));        // 4 MB
  __bf16* o0T    = (__bf16*)(ws + (512 << 10) + (24 << 20));       // 2 MB  (~26.5 MB total)

  k_invdeg<<<BB * EE * NN / 8, 256, 0, stream>>>(adj, invdeg);
  k_lin1<<<BB * NN * DD0 / 256, 256, 0, stream>>>(x, W1, b1, h0, h0T);
  k_agg<4><<<BB * (NN / 16) / 8, 256, 0, stream>>>(adj, invdeg, h0, h0T, hidden);
  k_lin2<<<BB * NN * DD1 / 256, 256, 0, stream>>>(hidden, W2, b2, o0, o0T);
  k_agg<2><<<BB * (NN / 16) / 8, 256, 0, stream>>>(adj, invdeg, o0, o0T, out);
}
